// GNN_84378927497676
// MI455X (gfx1250) — compile-verified, matrix operand order checked by reading, not run on hardware
//
#include <hip/hip_runtime.h>
#include <hip/hip_bf16.h>

// ---------------- problem constants (match reference) ----------------
#define N_NODES 50000
#define N_EDGES 800000
#define ET      (N_NODES + N_EDGES)   // 850000 edges incl. self loops
#define DIN     64
#define HID     64
#define HEADS   4
#define F1      (HEADS * HID)         // 256
#define NGRAPH  500
#define NOUT    2

typedef __attribute__((ext_vector_type(16))) __bf16 v16bf;
typedef __attribute__((ext_vector_type(8)))  float  v8f;

__device__ __forceinline__ int edge_src(const int* ei, int e) {
    return e < N_EDGES ? ei[e] : (e - N_EDGES);
}
__device__ __forceinline__ int edge_dst(const int* ei, int e) {
    return e < N_EDGES ? ei[N_EDGES + e] : (e - N_EDGES);
}

// order-preserving float -> uint encoding for atomicMax-based segment max
__device__ __forceinline__ unsigned enc_f32(float f) {
    unsigned u = __float_as_uint(f);
    return (u & 0x80000000u) ? ~u : (u | 0x80000000u);
}
__device__ __forceinline__ float dec_f32(unsigned u) {
    unsigned b = (u & 0x80000000u) ? (u & 0x7fffffffu) : ~u;
    return __uint_as_float(b);
}

__global__ void zero_f32(float* p, long n) {
    long i = (long)blockIdx.x * blockDim.x + threadIdx.x;
    if (i < n) p[i] = 0.f;
}

// ---------------- WMMA GEMM: H[M,NC] = X[M,K] @ W[K,NC] -------------------
// bf16 inputs / f32 accumulate on v_wmma_f32_16x16x32_bf16.
// W is staged once per block into LDS *pre-swizzled into B-fragment-major
// layout*: fragment f = kstep*(NC/16)+coltile, then per lane the 16 bf16
// fragment elements are contiguous and 32B-aligned. A B-fragment load is one
// 32-byte vector LDS read (2x ds_load_b128) instead of 16x ds_load_u16.
// Fragment element mapping follows the ISA 16-bit B layout:
//   lanes 0-15  (col = lane):    elems = K [0..7]  then [16..23]  (of kstep)
//   lanes 16-31 (col = lane-16): elems = K [8..15] then [24..31]
// Each wave register-blocks 4 col tiles (16x64 out), reusing its A fragment.
template <int K, int NC>
__global__ void wmma_gemm_bf16(const float* __restrict__ X,
                               const float* __restrict__ W,
                               float* __restrict__ Hout, int M) {
    constexpr int NT = 4;                    // col tiles per wave (64 cols)
    constexpr int NGROUP = NC / (16 * NT);   // col groups per row tile
    constexpr int NCT = NC / 16;             // col tiles total
    __shared__ __align__(32) __bf16 Wl[K * NC];

    // cooperative stage of W into LDS, f32 -> bf16, fragment-major swizzle
    for (int idx = threadIdx.x; idx < K * NC; idx += blockDim.x) {
        int k = idx / NC, c = idx - (idx / NC) * NC;
        int ks   = k >> 5;                   // 32-wide k step
        int kr   = k & 31;
        int half = (kr >> 3) & 1;            // 1 for K in [8..15] | [24..31]
        int i    = (kr & 7) + ((kr & 16) ? 8 : 0);
        int lane = (half << 4) | (c & 15);
        int f    = ks * NCT + (c >> 4);
        Wl[(((f << 5) + lane) << 4) + i] = (__bf16)W[idx];
    }
    __syncthreads();

    const int wave = threadIdx.x >> 5;
    const int lane = threadIdx.x & 31;
    const int job  = blockIdx.x * (blockDim.x >> 5) + wave;
    const int total = (M >> 4) * NGROUP;
    if (job >= total) return;                // wave-uniform: EXEC all-ones
    const int rt = job / NGROUP;             // row tile
    const int cg = job - rt * NGROUP;        // col group (16*NT cols)

    const int half = lane >> 4;              // 0: K 0-7/16-23, 1: K 8-15/24-31
    const int l15  = lane & 15;
    const int row  = (rt << 4) + l15;

    v8f acc[NT] = {};
#pragma unroll
    for (int k0 = 0; k0 < K; k0 += 32) {
        const int ks = k0 >> 5;
        const int kb = k0 + (half ? 8 : 0);
        v16bf a;
        const float* xr = X + (size_t)row * K + kb;
#pragma unroll
        for (int i = 0; i < 8; ++i) {
            a[i]     = (__bf16)xr[i];
            a[i + 8] = (__bf16)xr[i + 16];
        }
#pragma unroll
        for (int t = 0; t < NT; ++t) {
            const int f = ks * NCT + cg * NT + t;
            const v16bf b = *(const v16bf*)&Wl[((f << 5) + lane) << 4];
            acc[t] = __builtin_amdgcn_wmma_f32_16x16x32_bf16(
                         false, a, false, b, (short)0, acc[t], false, false);
        }
    }
    const int mbase = (rt << 4) + (half ? 8 : 0);
#pragma unroll
    for (int t = 0; t < NT; ++t) {
        const int col = cg * (16 * NT) + t * 16 + l15;
        float* o = Hout + (size_t)mbase * NC + col;
#pragma unroll
        for (int r = 0; r < 8; ++r) o[(size_t)r * NC] = acc[t][r];
    }
}

// ---------------- GAT pieces ----------------
__global__ void gat_scores(const float* __restrict__ H,
                           const float* __restrict__ att_s,
                           const float* __restrict__ att_d,
                           float* __restrict__ a_s, float* __restrict__ a_d) {
    int idx = blockIdx.x * blockDim.x + threadIdx.x;   // n*HEADS + h
    if (idx >= N_NODES * HEADS) return;
    int n = idx >> 2, h = idx & 3;
    const float* hp = H + (size_t)n * F1 + h * HID;
    const float* as = att_s + h * HID;
    const float* ad = att_d + h * HID;
    float s = 0.f, d = 0.f;
#pragma unroll 8
    for (int c = 0; c < HID; ++c) { s += hp[c] * as[c]; d += hp[c] * ad[c]; }
    a_s[idx] = s; a_d[idx] = d;
}

__global__ void gat_edge_max(const float* __restrict__ a_s,
                             const float* __restrict__ a_d,
                             const int* __restrict__ ei,
                             unsigned* __restrict__ m_enc) {
    int idx = blockIdx.x * blockDim.x + threadIdx.x;   // e*HEADS + h
    if (idx >= ET * HEADS) return;
    int e = idx >> 2, h = idx & 3;
    int s = edge_src(ei, e), d = edge_dst(ei, e);
    float v = a_s[s * 4 + h] + a_d[d * 4 + h];
    v = v > 0.f ? v : 0.2f * v;                        // leaky relu
    atomicMax(&m_enc[d * 4 + h], enc_f32(v));
}

__global__ void gat_edge_exp(const float* __restrict__ a_s,
                             const float* __restrict__ a_d,
                             const int* __restrict__ ei,
                             const unsigned* __restrict__ m_enc,
                             float* __restrict__ denom,
                             float* __restrict__ ex) {
    int idx = blockIdx.x * blockDim.x + threadIdx.x;
    if (idx >= ET * HEADS) return;
    int e = idx >> 2, h = idx & 3;
    int s = edge_src(ei, e), d = edge_dst(ei, e);
    float v = a_s[s * 4 + h] + a_d[d * 4 + h];
    v = v > 0.f ? v : 0.2f * v;
    float m = dec_f32(m_enc[d * 4 + h]);
    float x = __expf(v - m);
    ex[idx] = x;
    atomicAdd(&denom[d * 4 + h], x);
}

// one block (256 thr) per edge: thread t covers (head t/64, chan t%64)
__global__ void gat_edge_agg(const float* __restrict__ H,
                             const float* __restrict__ ex,
                             const float* __restrict__ denom,
                             const int* __restrict__ ei,
                             float* __restrict__ agg) {
    int e = blockIdx.x;
    int t = threadIdx.x;
    int h = t >> 6;
    int s = edge_src(ei, e), d = edge_dst(ei, e);
    float alpha = ex[e * 4 + h] / (denom[d * 4 + h] + 1e-16f);
    float v = H[(size_t)s * F1 + t] * alpha;
    atomicAdd(&agg[(size_t)d * F1 + t], v);            // agg fits in 192MB L2
}

__global__ void gat_head_mean(const float* __restrict__ agg,
                              const float* __restrict__ bias,
                              float* __restrict__ out) {
    int idx = blockIdx.x * blockDim.x + threadIdx.x;   // n*HID + c
    if (idx >= N_NODES * HID) return;
    int n = idx >> 6, c = idx & 63;
    const float* a = agg + (size_t)n * F1 + c;
    out[idx] = 0.25f * (a[0] + a[64] + a[128] + a[192]) + bias[c];
}

// ---------------- batch norm (block per channel, LDS tree reduce) ----------
__global__ void bn_stats(const float* __restrict__ h,
                         float* __restrict__ mean, float* __restrict__ invstd) {
    __shared__ float ssum[256];
    __shared__ float ssq[256];
    int c = blockIdx.x, t = threadIdx.x;
    float s = 0.f, q = 0.f;
    for (int i = t; i < N_NODES; i += 256) {
        float v = h[(size_t)i * HID + c];
        s += v; q += v * v;
    }
    ssum[t] = s; ssq[t] = q;
    __syncthreads();
    for (int st = 128; st > 0; st >>= 1) {
        if (t < st) { ssum[t] += ssum[t + st]; ssq[t] += ssq[t + st]; }
        __syncthreads();
    }
    if (t == 0) {
        float mu = ssum[0] / (float)N_NODES;
        float var = ssq[0] / (float)N_NODES - mu * mu;
        mean[c] = mu;
        invstd[c] = rsqrtf(var + 1e-5f);
    }
}

__global__ void bn_apply_relu(float* __restrict__ h,
                              const float* __restrict__ mean,
                              const float* __restrict__ invstd,
                              const float* __restrict__ gamma,
                              const float* __restrict__ beta) {
    int idx = blockIdx.x * blockDim.x + threadIdx.x;
    if (idx >= N_NODES * HID) return;
    int c = idx & 63;
    float v = (h[idx] - mean[c]) * invstd[c] * gamma[c] + beta[c];
    h[idx] = v > 0.f ? v : 0.f;
}

// ---------------- GCN ----------------
__global__ void gcn_deg(const int* __restrict__ ei, float* __restrict__ deg) {
    int e = blockIdx.x * blockDim.x + threadIdx.x;
    if (e >= ET) return;
    atomicAdd(&deg[edge_dst(ei, e)], 1.0f);
}

__global__ void gcn_scatter(const float* __restrict__ hg,
                            const float* __restrict__ deg,
                            const int* __restrict__ ei,
                            float* __restrict__ gout) {
    long idx = (long)blockIdx.x * blockDim.x + threadIdx.x;
    if (idx >= (long)ET * HID) return;
    int e = (int)(idx >> 6), c = (int)(idx & 63);
    int s = edge_src(ei, e), d = edge_dst(ei, e);
    float norm = rsqrtf(deg[s]) * rsqrtf(deg[d]);
    atomicAdd(&gout[(size_t)d * HID + c], hg[(size_t)s * HID + c] * norm);
}

__global__ void bias_relu(const float* __restrict__ in,
                          const float* __restrict__ b,
                          float* __restrict__ out) {
    int idx = blockIdx.x * blockDim.x + threadIdx.x;
    if (idx >= N_NODES * HID) return;
    float v = in[idx] + b[idx & 63];
    out[idx] = v > 0.f ? v : 0.f;
}

// ---------------- pooling + MLP head ----------------
__global__ void pool_sum(const float* __restrict__ h,
                         const int* __restrict__ batch,
                         float* __restrict__ psum, float* __restrict__ pcnt) {
    int idx = blockIdx.x * blockDim.x + threadIdx.x;
    if (idx >= N_NODES * HID) return;
    int n = idx >> 6, c = idx & 63;
    int g = batch[n];
    atomicAdd(&psum[(size_t)g * HID + c], h[idx]);
    if (c == 0) atomicAdd(&pcnt[g], 1.0f);
}

__global__ void mlp_fc1(const float* __restrict__ psum,
                        const float* __restrict__ pcnt,
                        const float* __restrict__ w,
                        const float* __restrict__ b,
                        float* __restrict__ out) {
    int idx = blockIdx.x * blockDim.x + threadIdx.x;   // g*HID + j
    if (idx >= NGRAPH * HID) return;
    int g = idx >> 6, j = idx & 63;
    float cnt = pcnt[g]; cnt = cnt > 1.f ? cnt : 1.f;
    float inv = 1.f / cnt;
    float s = b[j];
#pragma unroll 8
    for (int c = 0; c < HID; ++c)
        s += psum[(size_t)g * HID + c] * inv * w[c * HID + j];
    out[idx] = s > 0.f ? s : 0.f;
}

__global__ void mlp_fc2(const float* __restrict__ h1,
                        const float* __restrict__ w,
                        const float* __restrict__ b,
                        float* __restrict__ out) {
    int idx = blockIdx.x * blockDim.x + threadIdx.x;   // g*NOUT + o
    if (idx >= NGRAPH * NOUT) return;
    int g = idx >> 1, o = idx & 1;
    float s = b[o];
#pragma unroll 8
    for (int j = 0; j < HID; ++j)
        s += h1[(size_t)g * HID + j] * w[j * NOUT + o];
    out[idx] = s;
}

// ---------------- host orchestration ----------------
extern "C" void kernel_launch(void* const* d_in, const int* in_sizes, int n_in,
                              void* d_out, int out_size, void* d_ws, size_t ws_size,
                              hipStream_t stream) {
    const float* x    = (const float*)d_in[0];
    const int*   ei   = (const int*)  d_in[1];
    const int*   batch= (const int*)  d_in[2];
    const float* W1   = (const float*)d_in[3];
    const float* as1  = (const float*)d_in[4];
    const float* ad1  = (const float*)d_in[5];
    const float* b1   = (const float*)d_in[6];
    const float* g1   = (const float*)d_in[7];
    const float* be1  = (const float*)d_in[8];
    const float* W2   = (const float*)d_in[9];
    const float* as2  = (const float*)d_in[10];
    const float* ad2  = (const float*)d_in[11];
    const float* b2   = (const float*)d_in[12];
    const float* g2   = (const float*)d_in[13];
    const float* be2  = (const float*)d_in[14];
    const float* Wg   = (const float*)d_in[15];
    const float* bg   = (const float*)d_in[16];
    const float* fc1w = (const float*)d_in[17];
    const float* fc1b = (const float*)d_in[18];
    const float* fc2w = (const float*)d_in[19];
    const float* fc2b = (const float*)d_in[20];
    float* out = (float*)d_out;

    // workspace carve-up (256B aligned)
    char* ws = (char*)d_ws;
    size_t off = 0;
    auto alloc = [&](size_t bytes) -> char* {
        char* p = ws + off;
        off = (off + bytes + 255) & ~(size_t)255;
        return p;
    };
    float*    h256  = (float*)   alloc((size_t)N_NODES * F1 * 4);
    float*    agg   = (float*)   alloc((size_t)N_NODES * F1 * 4);
    float*    hsm   = (float*)   alloc((size_t)N_NODES * HID * 4);
    float*    a_s   = (float*)   alloc((size_t)N_NODES * HEADS * 4);
    float*    a_d   = (float*)   alloc((size_t)N_NODES * HEADS * 4);
    unsigned* menc  = (unsigned*)alloc((size_t)N_NODES * HEADS * 4);
    float*    denom = (float*)   alloc((size_t)N_NODES * HEADS * 4);
    float*    ex    = (float*)   alloc((size_t)ET * HEADS * 4);
    float*    deg   = (float*)   alloc((size_t)N_NODES * 4);
    float*    bnm   = (float*)   alloc(64 * 4);
    float*    bni   = (float*)   alloc(64 * 4);
    float*    psum  = (float*)   alloc((size_t)NGRAPH * HID * 4);
    float*    pcnt  = (float*)   alloc((size_t)NGRAPH * 4);
    float*    mlp1  = (float*)   alloc((size_t)NGRAPH * HID * 4);
    float*    hg    = h256;   // GCN GEMM output reuses GAT feature buffer
    float*    gout  = agg;    // GCN scatter target reuses aggregation buffer

    auto cdiv = [](long a, long b) { return (int)((a + b - 1) / b); };

    auto run_gat = [&](const float* Xin, const float* Wl,
                       const float* attS, const float* attD, const float* bias) {
        zero_f32<<<cdiv((long)N_NODES * F1, 256), 256, 0, stream>>>(agg, (long)N_NODES * F1);
        zero_f32<<<cdiv((long)N_NODES * HEADS, 256), 256, 0, stream>>>((float*)menc, (long)N_NODES * HEADS);
        zero_f32<<<cdiv((long)N_NODES * HEADS, 256), 256, 0, stream>>>(denom, (long)N_NODES * HEADS);
        // GEMM: [N,64] @ [64,256]; wave job = 16 rows x 64 cols
        wmma_gemm_bf16<64, F1><<<cdiv((long)(N_NODES / 16) * (F1 / 64), 8), 256, 0, stream>>>(
            Xin, Wl, h256, N_NODES);
        gat_scores<<<cdiv((long)N_NODES * HEADS, 256), 256, 0, stream>>>(h256, attS, attD, a_s, a_d);
        gat_edge_max<<<cdiv((long)ET * HEADS, 256), 256, 0, stream>>>(a_s, a_d, ei, menc);
        gat_edge_exp<<<cdiv((long)ET * HEADS, 256), 256, 0, stream>>>(a_s, a_d, ei, menc, denom, ex);
        gat_edge_agg<<<ET, 256, 0, stream>>>(h256, ex, denom, ei, agg);
        gat_head_mean<<<cdiv((long)N_NODES * HID, 256), 256, 0, stream>>>(agg, bias, hsm);
    };
    auto run_bn = [&](const float* gamma, const float* beta) {
        bn_stats<<<64, 256, 0, stream>>>(hsm, bnm, bni);
        bn_apply_relu<<<cdiv((long)N_NODES * HID, 256), 256, 0, stream>>>(hsm, bnm, bni, gamma, beta);
    };

    // GAT1 + BN + ReLU
    run_gat(x, W1, as1, ad1, b1);
    run_bn(g1, be1);
    // GAT2 + BN + ReLU
    run_gat(hsm, W2, as2, ad2, b2);
    run_bn(g2, be2);

    // GCN: h @ Wg -> normalized scatter -> bias+relu
    wmma_gemm_bf16<64, HID><<<cdiv((long)(N_NODES / 16) * (HID / 64), 8), 256, 0, stream>>>(
        hsm, Wg, hg, N_NODES);
    zero_f32<<<cdiv((long)N_NODES * HID, 256), 256, 0, stream>>>(gout, (long)N_NODES * HID);
    zero_f32<<<cdiv((long)N_NODES, 256), 256, 0, stream>>>(deg, (long)N_NODES);
    gcn_deg<<<cdiv((long)ET, 256), 256, 0, stream>>>(ei, deg);
    gcn_scatter<<<cdiv((long)ET * HID, 256), 256, 0, stream>>>(hg, deg, ei, gout);
    bias_relu<<<cdiv((long)N_NODES * HID, 256), 256, 0, stream>>>(gout, bg, hsm);

    // global mean pool + MLP head
    zero_f32<<<cdiv((long)NGRAPH * HID, 256), 256, 0, stream>>>(psum, (long)NGRAPH * HID);
    zero_f32<<<cdiv((long)NGRAPH, 256), 256, 0, stream>>>(pcnt, (long)NGRAPH);
    pool_sum<<<cdiv((long)N_NODES * HID, 256), 256, 0, stream>>>(hsm, batch, psum, pcnt);
    mlp_fc1<<<cdiv((long)NGRAPH * HID, 256), 256, 0, stream>>>(psum, pcnt, fc1w, fc1b, mlp1);
    mlp_fc2<<<cdiv((long)NGRAPH * NOUT, 256), 256, 0, stream>>>(mlp1, fc2w, fc2b, out);
}
